// AttentionLayer_48644799594708
// MI455X (gfx1250) — matrix operand order repaired and checked by reading, not compiled, not run on hardware
//
#include <hip/hip_runtime.h>
#include <hip/hip_bf16.h>

#define B_  2048
#define T_  200
#define D_  128
#define H1_ 128
#define H2_ 64
#define TPAD 208      // 13 * 16
#define MTILES 13
#define MASK_PAD -4294967295.0f

#define SA_STRIDE 520 // ushorts; 1040B rows -> 16 lanes spread over all 64 LDS banks
#define SH_STRIDE 136 // ushorts; 272B rows  -> same conflict-free spread

typedef __attribute__((ext_vector_type(16))) __bf16 v16bf;
typedef __attribute__((ext_vector_type(8)))  float  v8f;

union Frag16 { uint4 q[2]; v16bf v; };

// Native bf16 convert (RNE) + pack two into one dword for 32-bit LDS stores.
__device__ __forceinline__ unsigned short bfbits(float x) {
    union { __bf16 h; unsigned short s; } u; u.h = (__bf16)x; return u.s;
}
__device__ __forceinline__ unsigned pk2(float a, float b) {
    return (unsigned)bfbits(a) | ((unsigned)bfbits(b) << 16);
}

// One-time weight transpose + fp32 -> bf16 conversion into workspace.
__global__ void din_prep_weights(const float* __restrict__ W0,
                                 const float* __restrict__ W1,
                                 unsigned short* __restrict__ w0t,
                                 unsigned short* __restrict__ w1t) {
    int i = blockIdx.x * 256 + threadIdx.x;
    if (i < 4 * D_ * H1_) {            // W0: [512,128] -> W0^T bf16 [128][512]
        int k = i / H1_, n = i % H1_;
        w0t[n * (4 * D_) + k] = bfbits(W0[i]);
    }
    if (i < H1_ * H2_) {               // W1: [128,64] -> W1^T bf16 [64][128]
        int h = i / H2_, g = i % H2_;
        w1t[g * H1_ + h] = bfbits(W1[i]);
    }
}

__global__ __launch_bounds__(256) void din_attention_kernel(
    const float* __restrict__ query, const float* __restrict__ key,
    const float* __restrict__ val,   const int*   __restrict__ mask,
    const unsigned short* __restrict__ w0t, const float* __restrict__ b0,
    const float* __restrict__ a0,    const unsigned short* __restrict__ w1t,
    const float* __restrict__ b1,    const float* __restrict__ a1,
    const float* __restrict__ Wout,  const float* __restrict__ bout,
    float* __restrict__ out)
{
    __shared__ unsigned short sA[16 * SA_STRIDE];   // feat tile, bf16
    __shared__ unsigned short sH0[16 * SH_STRIDE];  // hidden-1 tile, bf16
    __shared__ float sQ[D_];
    __shared__ float sLogits[TPAD];
    __shared__ float sRed[256];
    __shared__ float sW[8];                         // per-wave reduction slots

    const int tid  = threadIdx.x;
    const int b    = blockIdx.x;
    const int w    = tid >> 5;       // wave 0..7  (wave32)
    const int lane = tid & 31;
    const int ln   = lane & 15;
    const int kh   = lane >> 4;      // K-half select per ISA 16-bit A/B layout

    if (tid < D_)   sQ[tid]      = query[(size_t)b * D_ + tid];
    if (tid < TPAD) sLogits[tid] = 0.0f;

    // ---- Preload all GEMM1 B-fragments (W0^T columns) into VGPRs ----
    Frag16 B0[16];
    {
        const int col = (w << 4) + ln;                 // H1 column 0..127
        #pragma unroll
        for (int ks = 0; ks < 16; ++ks) {
            const unsigned short* p = w0t + col * 512 + ks * 32 + kh * 8;
            B0[ks].q[0] = *reinterpret_cast<const uint4*>(p);
            B0[ks].q[1] = *reinterpret_cast<const uint4*>(p + 16);
        }
    }
    // ---- Preload GEMM2 B-fragments (W1^T columns); used by waves 0..3 ----
    Frag16 B1[4];
    {
        const int col = ((w & 3) << 4) + ln;           // H2 column 0..63
        #pragma unroll
        for (int ks = 0; ks < 4; ++ks) {
            const unsigned short* p = w1t + col * 128 + ks * 32 + kh * 8;
            B1[ks].q[0] = *reinterpret_cast<const uint4*>(p);
            B1[ks].q[1] = *reinterpret_cast<const uint4*>(p + 16);
        }
    }
    __syncthreads();

    const size_t keyBase = (size_t)b * T_ * D_;
    const size_t valBase = (size_t)b * T_ * D_;

    for (int mt = 0; mt < MTILES; ++mt) {
        const int t0 = mt * 16;
        __syncthreads();   // previous tile's GEMM2 reads of sH0 must be done

        // ---- build feat tile [16 x 512] bf16: [q | k | q-k | q*k] ----
        #pragma unroll
        for (int i = 0; i < 4; ++i) {
            int idx = tid + i * 256;        // 0..1023 -> (t_local, d-pair)
            int tl  = idx >> 6;             // 0..15
            int d   = (idx & 63) << 1;      // even d
            int t   = t0 + tl;
            float q0 = sQ[d], q1 = sQ[d + 1];
            float k0 = 0.0f, k1 = 0.0f;
            if (t < T_) {
                float2 kv = *reinterpret_cast<const float2*>(
                                &key[keyBase + (size_t)t * D_ + d]);
                k0 = kv.x; k1 = kv.y;
            }
            unsigned short* row = sA + tl * SA_STRIDE;
            *reinterpret_cast<unsigned*>(&row[d])       = pk2(q0, q1);
            *reinterpret_cast<unsigned*>(&row[128 + d]) = pk2(k0, k1);
            *reinterpret_cast<unsigned*>(&row[256 + d]) = pk2(q0 - k0, q1 - k1);
            *reinterpret_cast<unsigned*>(&row[384 + d]) = pk2(q0 * k0, q1 * k1);
        }
        __syncthreads();

        // ---- prefetch: next tile's key lines + progressive val warm-up ----
        {
            const int nt0 = t0 + 16;
            if (tid < 64 && nt0 < T_) {          // 16 rows x 512B = 64 x 128B lines
                int r = tid >> 2, c = tid & 3;
                int t = nt0 + r;
                if (t < T_)
                    __builtin_prefetch(&key[keyBase + (size_t)t * D_ + c * 32], 0, 3);
            }
            if (tid >= 64 && tid < 128) {        // val: 800 lines over 13 tiles
                int li = mt * 64 + (tid - 64);
                if (li < (T_ * D_) / 32)
                    __builtin_prefetch(&val[valBase + (size_t)li * 32], 0, 3);
            }
        }

        // ---- GEMM1: feat[16x512] x W0[512x16-per-wave] ----
        // 3-deep software pipeline on A fragments: at wmma(ks), loads for
        // ks+1/ks+2 are in flight -> s_wait_dscnt <= 4 instead of 0.
        v8f acc = {};
        {
            const unsigned short* arow = sA + ln * SA_STRIDE + kh * 8;
            Frag16 af[3];
            #pragma unroll
            for (int p = 0; p < 3; ++p) {
                af[p].q[0] = *reinterpret_cast<const uint4*>(arow + p * 32);
                af[p].q[1] = *reinterpret_cast<const uint4*>(arow + p * 32 + 16);
            }
            #pragma unroll
            for (int ks = 0; ks < 16; ++ks) {
                acc = __builtin_amdgcn_wmma_f32_16x16x32_bf16(
                        false, af[ks % 3].v, false, B0[ks].v, (short)0, acc,
                        false, false);
                if (ks + 3 < 16) {
                    af[ks % 3].q[0] =
                        *reinterpret_cast<const uint4*>(arow + (ks + 3) * 32);
                    af[ks % 3].q[1] =
                        *reinterpret_cast<const uint4*>(arow + (ks + 3) * 32 + 16);
                }
            }
        }

        // ---- epilogue: + b0, PReLU(a0[t,g]) -> sH0 bf16 ----
        {
            const int g = (w << 4) + ln;
            const float bias = b0[g];
            #pragma unroll
            for (int r = 0; r < 8; ++r) {
                int m = r + (kh << 3);                   // C layout: M = r + 8*half
                int t = t0 + m; if (t >= T_) t = T_ - 1; // clamp for alpha read
                float h  = acc[r] + bias;
                float al = a0[t * H1_ + g];
                h = (h > 0.0f) ? h : al * h;
                sH0[m * SH_STRIDE + g] = bfbits(h);
            }
        }
        __syncthreads();

        // ---- GEMM2 (waves 0..3): h0[16x128] x W1[128x16], then logits ----
        if (w < 4) {
            const unsigned short* arow = sH0 + ln * SH_STRIDE + kh * 8;
            Frag16 af[4];
            #pragma unroll
            for (int ks = 0; ks < 4; ++ks) {             // all loads up-front
                af[ks].q[0] = *reinterpret_cast<const uint4*>(arow + ks * 32);
                af[ks].q[1] = *reinterpret_cast<const uint4*>(arow + ks * 32 + 16);
            }
            v8f acc2 = {};
            #pragma unroll
            for (int ks = 0; ks < 4; ++ks) {
                acc2 = __builtin_amdgcn_wmma_f32_16x16x32_bf16(
                        false, af[ks].v, false, B1[ks].v, (short)0, acc2,
                        false, false);
            }
            const int g = (w << 4) + ln;
            const float bias = b1[g];
            const float wo   = Wout[g];
            #pragma unroll
            for (int r = 0; r < 8; ++r) {
                int m = r + (kh << 3);
                int t = t0 + m;
                if (t < T_) {
                    float h  = acc2[r] + bias;
                    float al = a1[t * H2_ + g];
                    h = (h > 0.0f) ? h : al * h;
                    atomicAdd(&sLogits[t], h * wo);      // ds_add_f32
                }
            }
        }
    }
    __syncthreads();

    // ---- masked softmax over T=200 (wave-shuffle reductions) ----
    const float bo = bout[0];
    float l = -3.0e38f;
    if (tid < T_) {
        float raw = sLogits[tid] + bo;
        l = (mask[(size_t)b * T_ + tid] == 0) ? MASK_PAD : raw;
    }
    {   // block max
        float v = l;
        #pragma unroll
        for (int off = 16; off > 0; off >>= 1)
            v = fmaxf(v, __shfl_xor(v, off, 32));
        if (lane == 0) sW[w] = v;
    }
    __syncthreads();
    float mx = sW[0];
    #pragma unroll
    for (int i = 1; i < 8; ++i) mx = fmaxf(mx, sW[i]);

    float e = 0.0f;
    if (tid < T_) e = __expf(l - mx);
    __syncthreads();               // sW reuse
    {   // block sum
        float v = e;
        #pragma unroll
        for (int off = 16; off > 0; off >>= 1)
            v += __shfl_xor(v, off, 32);
        if (lane == 0) sW[w] = v;
    }
    __syncthreads();
    float sum = sW[0];
    #pragma unroll
    for (int i = 1; i < 8; ++i) sum += sW[i];
    const float inv = 1.0f / sum;

    if (tid < T_) sLogits[tid] = e * inv;
    __syncthreads();

    // ---- out[b,d] = sum_t attn[t] * val[b,t,d]  (2-way T split) ----
    {
        const int d      = tid & 127;
        const int tstart = (tid >> 7) * 100;
        const float* vb  = val + valBase;
        float p = 0.0f;
        #pragma unroll 4
        for (int t = tstart; t < tstart + 100; ++t)
            p = fmaf(sLogits[t], vb[(size_t)t * D_ + d], p);
        sRed[tid] = p;
        __syncthreads();
        if (tid < 128) out[(size_t)b * D_ + tid] = sRed[tid] + sRed[tid + 128];
    }
}

extern "C" void kernel_launch(void* const* d_in, const int* in_sizes, int n_in,
                              void* d_out, int out_size, void* d_ws, size_t ws_size,
                              hipStream_t stream) {
    const float* query = (const float*)d_in[0];
    const float* key   = (const float*)d_in[1];
    const float* val   = (const float*)d_in[2];
    const int*   mask  = (const int*)d_in[3];
    const float* W0    = (const float*)d_in[4];
    const float* b0    = (const float*)d_in[5];
    const float* a0    = (const float*)d_in[6];
    const float* W1    = (const float*)d_in[7];
    const float* b1    = (const float*)d_in[8];
    const float* a1    = (const float*)d_in[9];
    const float* Wout  = (const float*)d_in[10];
    const float* bout  = (const float*)d_in[11];
    float* out = (float*)d_out;

    unsigned short* w0t = (unsigned short*)d_ws;        // 512*128 bf16 = 128 KiB
    unsigned short* w1t = w0t + 4 * D_ * H1_;           // 128*64  bf16 = 16 KiB

    din_prep_weights<<<(4 * D_ * H1_ + 255) / 256, 256, 0, stream>>>(W0, W1, w0t, w1t);
    din_attention_kernel<<<B_, 256, 0, stream>>>(query, key, val, mask,
                                                 w0t, b0, a0, w1t, b1, a1,
                                                 Wout, bout, out);
}
